// RoPEMultiHeadAttention_88046829568782
// MI455X (gfx1250) — matrix operand order repaired and checked by reading, not compiled, not run on hardware
//
#include <hip/hip_runtime.h>
#include <hip/hip_bf16.h>

// ---------------------------------------------------------------------------
// RoPE multi-head attention for MI455X (gfx1250, wave32, WMMA).
//
// Roofline: ~69 GFLOP of GEMM vs ~50 MB of unique traffic -> compute bound.
// => all matmuls on v_wmma_f32_16x16x32_f16 (f16 in, f32 accumulate).
// Unit-variance data makes f16 (~1e-3 rel err) safe; fp32 accumulation keeps
// the K=1024 / T=2048 reductions accurate.
//
// Data-movement choices (L2 = 192 MB, HBM 23.3 TB/s):
//  - GEMM blocks: 128x64 tile, wave owns an M sub-tile (A fragment loaded
//    ONCE per K-chunk from global), B fragments re-read from LDS (cheap).
//  - W staged to LDS as f16 transposed so B-fragments are 2x ds_load_b128.
//  - V stored transposed, q/k stored f16 [B,H,T,D]: every WMMA fragment in
//    the attention kernel is built from contiguous 16B loads.
//  - softmax scale 1/sqrt(D) folded into q at pack time.
//
// Workspace layout (d_ws), ~84 MB:
//   qkv  fp32 [4096 x 3072]               50,331,648 B
//   qh   f16  [B,H,T,D] (pre-scaled)       8,388,608 B
//   kh   f16  [B,H,T,D]                    8,388,608 B
//   vt   f16  [B,H,D,T] (V transposed)     8,388,608 B
//   attn f16  [4096 x 1024]                8,388,608 B
// ---------------------------------------------------------------------------

typedef _Float16 v2h  __attribute__((ext_vector_type(2)));
typedef _Float16 v8h  __attribute__((ext_vector_type(8)));
typedef _Float16 v16h __attribute__((ext_vector_type(16)));
typedef float    v8f  __attribute__((ext_vector_type(8)));

static constexpr int B_  = 2;
static constexpr int T_  = 2048;
static constexpr int C_  = 1024;
static constexpr int H_  = 16;
static constexpr int D_  = 64;
static constexpr int M_  = B_ * T_;     // 4096 rows
static constexpr int SP_ = T_ + 8;      // padded S row stride (f16), 16B-aligned

__device__ __forceinline__ v16h cat16(v8h lo, v8h hi) {
  return __builtin_shufflevector(lo, hi, 0,1,2,3,4,5,6,7,8,9,10,11,12,13,14,15);
}

// B-matrix / V fragment: 8 f16 at p (K = k0..k0+7) and 8 f16 at p+16 (K = k0+16..k0+23): two b128 loads.
__device__ __forceinline__ v16h load16_contig(const _Float16* p) {
  return cat16(*(const v8h*)p, *(const v8h*)(p + 16));
}

// A-matrix fragment: 8 f16 at p (K = k0..k0+7) and 8 f16 at p+16 (K = k0+16..k0+23).
__device__ __forceinline__ v16h load16_gap(const _Float16* p) {
  return cat16(*(const v8h*)p, *(const v8h*)(p + 16));
}

// A-matrix fragment from fp32 source, converted to f16. float4 loads -> b128.
__device__ __forceinline__ v16h load16_f32(const float* p) {
  float4 f0 = *(const float4*)(p);
  float4 f1 = *(const float4*)(p + 4);
  float4 f2 = *(const float4*)(p + 16);
  float4 f3 = *(const float4*)(p + 20);
  v16h r;
  r[0]=(_Float16)f0.x;  r[1]=(_Float16)f0.y;  r[2]=(_Float16)f0.z;  r[3]=(_Float16)f0.w;
  r[4]=(_Float16)f1.x;  r[5]=(_Float16)f1.y;  r[6]=(_Float16)f1.z;  r[7]=(_Float16)f1.w;
  r[8]=(_Float16)f2.x;  r[9]=(_Float16)f2.y;  r[10]=(_Float16)f2.z; r[11]=(_Float16)f2.w;
  r[12]=(_Float16)f3.x; r[13]=(_Float16)f3.y; r[14]=(_Float16)f3.z; r[15]=(_Float16)f3.w;
  return r;
}

__device__ __forceinline__ v8f wmma16(v16h a, v16h b, v8f c) {
  // (neg_a, A, neg_b, B, c_mod, C, reuse_a, reuse_b)
  return __builtin_amdgcn_wmma_f32_16x16x32_f16(false, a, false, b, (short)0, c,
                                                false, false);
}

// ---------------------------------------------------------------------------
// GEMM: Out[M,N] = A[M,K] * W[K,N] + bias[N]  (fp32 out).
// Block = 256 thr = 8 waves computing a 128x64 output tile.
// Wave w owns M sub-tile w: its A fragment is loaded from global exactly once
// per K-chunk; the 4 N sub-tile B-fragments come from LDS (re-reads hit the
// LDS, not L2). W is staged f16 TRANSPOSED [64 cols][32 k] so B-fragments are
// two contiguous ds_load_b128 (ISA B layout: lanes 0-15 K=0..7,16..23, lanes 16-31
// K=8..15,24..31, col = lane&15).
// ---------------------------------------------------------------------------
template <bool A_IS_F16>
__global__ void gemm_wmma_kernel(const void* __restrict__ Aptr,
                                 const float* __restrict__ W,
                                 const float* __restrict__ bias,
                                 float* __restrict__ Out,
                                 int K, int N) {
  __shared__ _Float16 ldsW[64 * 32];   // [col][k] transposed, 4 KB

  const int tid   = threadIdx.x;
  const int lane  = tid & 31;
  const int wave  = tid >> 5;          // 0..7 -> M sub-tile
  const int row16 = lane & 15;
  const bool hi   = lane >= 16;

  const int mblk = blockIdx.x * 128;
  const int nblk = blockIdx.y * 64;
  const int arow = mblk + wave * 16 + row16;

  v8f acc[4] = {};

  for (int k0 = 0; k0 < K; k0 += 32) {
    // --- stage W[k0..k0+32) x [nblk..nblk+64) into LDS f16, transposed.
    // 256 threads cover the 32x64 chunk in one pass: each thread loads a pair
    // of adjacent K rows x 4 cols (two float4) and stores four packed v2h.
    {
      const int pr = tid >> 4;                 // K row-pair index: 0..15
      const int c4 = (tid & 15) * 4;           // col group: 0..60
      const float* wp = W + (size_t)(k0 + 2 * pr) * N + nblk + c4;
      const float4 w0 = *(const float4*)wp;
      const float4 w1 = *(const float4*)(wp + N);
      *(v2h*)&ldsW[(c4 + 0) * 32 + 2 * pr] = v2h{(_Float16)w0.x, (_Float16)w1.x};
      *(v2h*)&ldsW[(c4 + 1) * 32 + 2 * pr] = v2h{(_Float16)w0.y, (_Float16)w1.y};
      *(v2h*)&ldsW[(c4 + 2) * 32 + 2 * pr] = v2h{(_Float16)w0.z, (_Float16)w1.z};
      *(v2h*)&ldsW[(c4 + 3) * 32 + 2 * pr] = v2h{(_Float16)w0.w, (_Float16)w1.w};
    }
    __syncthreads();

    // --- A fragment for this wave's M sub-tile: loaded once per K-chunk ---
    v16h a;
    if constexpr (A_IS_F16) {
      a = load16_gap((const _Float16*)Aptr + (size_t)arow * K + k0 + (hi ? 8 : 0));
    } else {
      a = load16_f32((const float*)Aptr + (size_t)arow * K + k0 + (hi ? 8 : 0));
    }

    // --- 4 N sub-tiles: B fragments from LDS, A reused ---
#pragma unroll
    for (int nt = 0; nt < 4; ++nt) {
      const int col = nt * 16 + row16;
      const v16h b = load16_contig(&ldsW[col * 32 + (hi ? 8 : 0)]);
      acc[nt] = wmma16(a, b, acc[nt]);
    }
    __syncthreads();
  }

#pragma unroll
  for (int nt = 0; nt < 4; ++nt) {
    const int n  = nblk + nt * 16 + row16;
    const float bv = bias[n];
#pragma unroll
    for (int e = 0; e < 8; ++e) {
      const int r = (hi ? 8 : 0) + e;          // C/D layout: rows split by half
      Out[(size_t)(mblk + wave * 16 + r) * N + n] = acc[nt][e] + bv;
    }
  }
}

// ---------------------------------------------------------------------------
// RoPE + repack: qkv fp32 [B,T,3,H,D] -> q (roped, pre-scaled by 1/sqrt(D)),
// k (roped) f16 [B,H,T,D], v f16 transposed [B,H,D,T].
// ---------------------------------------------------------------------------
__global__ void rope_pack_kernel(const float* __restrict__ qkv,
                                 _Float16* __restrict__ qh,
                                 _Float16* __restrict__ kh,
                                 _Float16* __restrict__ vt) {
  const int idx = blockIdx.x * 256 + threadIdx.x;   // 0 .. B*H*T*D-1 (4M)
  const int d = idx & 63;
  const int t = (idx >> 6) & 2047;
  const int h = (idx >> 17) & 15;
  const int b = idx >> 21;

  const size_t rb = ((size_t)(b * T_ + t)) * 3072 + h * 64;
  const float qv = qkv[rb + d];
  const float kv = qkv[rb + 1024 + d];
  const float vv = qkv[rb + 2048 + d];

  // rotate_half: out[d] = -x[2d+1] (d<32)  |  x[2(d-32)] (d>=32)
  const int   pid = (d < 32) ? (2 * d + 1) : (2 * (d - 32));
  const float sgn = (d < 32) ? -1.0f : 1.0f;
  const float qr = qkv[rb + pid] * sgn;
  const float kr = qkv[rb + 1024 + pid] * sgn;

  const int j = d & 31;                         // inv_freq index
  const float freq = __expf(-(float)j * (9.2103403719761836f / 32.0f)); // 10000^(-j/32)
  const float ang  = (float)t * freq;
  const float sn = __sinf(ang);
  const float cs = __cosf(ang);

  const size_t o = ((size_t)((b * H_ + h) * T_ + t)) * D_ + d;
  qh[o] = (_Float16)((qv * cs + qr * sn) * 0.125f);   // fold 1/sqrt(64)
  kh[o] = (_Float16)(kv * cs + kr * sn);
  vt[((size_t)((b * H_ + h) * D_ + d)) * T_ + t] = (_Float16)vv;
}

// ---------------------------------------------------------------------------
// Attention: one WG (128 thr = 4 waves) per (b, h, 16-row q-tile).
// Phase 1: S[16][2048(+8 pad)] = Q Kᵀ (scale pre-folded) via WMMA, f16 LDS.
// Phase 2: row-max (packed f16 v_pk_max) / exp / row-sum; each thread scans a
//          CONTIGUOUS 256-f16 segment -> ds_load_b128 / ds_store_b128.
// Phase 3: O = P @ V via WMMA (wave w owns d-tile w), divide by row-sum,
//          store attn as f16 [B,T,C].
// ---------------------------------------------------------------------------
__global__ void attn_wmma_kernel(const _Float16* __restrict__ qh,
                                 const _Float16* __restrict__ kh,
                                 const _Float16* __restrict__ vt,
                                 _Float16* __restrict__ attn) {
  extern __shared__ char smem[];
  _Float16* S      = (_Float16*)smem;                       // [16][SP_]
  float*    part   = (float*)(smem + 16 * SP_ * 2);         // [16][8]
  float*    rowmax = part + 16 * 8;                         // [16]
  float*    rowsum = rowmax + 16;                           // [16]

  const int tid   = threadIdx.x;
  const int lane  = tid & 31;
  const int wave  = tid >> 5;
  const int row16 = lane & 15;
  const bool hi   = lane >= 16;

  const int qt = blockIdx.x & 127;       // q tile
  const int bh = blockIdx.x >> 7;        // b*16+h
  const _Float16* qb = qh + (size_t)bh * T_ * D_;
  const _Float16* kb = kh + (size_t)bh * T_ * D_;
  const _Float16* vb = vt + (size_t)bh * D_ * T_;

  // Q fragments for this tile (d-chunks 0..31 and 32..63), reused all phase 1.
  const int trow = qt * 16 + row16;
  const v16h a0 = load16_gap(qb + (size_t)trow * D_ + 0  + (hi ? 8 : 0));
  const v16h a1 = load16_gap(qb + (size_t)trow * D_ + 32 + (hi ? 8 : 0));

  // ---- Phase 1: scores ----
  for (int kt = wave; kt < T_ / 16; kt += 4) {
    const int key = kt * 16 + row16;
    const v16h b0 = load16_contig(kb + (size_t)key * D_ + 0  + (hi ? 8 : 0));
    const v16h b1 = load16_contig(kb + (size_t)key * D_ + 32 + (hi ? 8 : 0));
    v8f c = {};
    c = wmma16(a0, b0, c);
    c = wmma16(a1, b1, c);
#pragma unroll
    for (int e = 0; e < 8; ++e) {
      const int r = (hi ? 8 : 0) + e;
      S[r * SP_ + kt * 16 + row16] = (_Float16)c[e];   // scale already in q
    }
  }
  __syncthreads();

  // ---- Phase 2: softmax; 8 threads per row, contiguous 256-f16 segments ----
  const int srow = tid >> 3, slane = tid & 7;
  _Float16* Srow = S + srow * SP_ + slane * 256;

  // packed f16 max scan (v_pk_max_num_f16), single f32 reduce at the end
  v8h vm = *(const v8h*)(Srow);
#pragma unroll 4
  for (int i = 1; i < 32; ++i) {
    const v8h v = *(const v8h*)(Srow + i * 8);
    vm = __builtin_elementwise_max(vm, v);
  }
  float m = (float)vm[0];
#pragma unroll
  for (int e = 1; e < 8; ++e) m = fmaxf(m, (float)vm[e]);
  part[srow * 8 + slane] = m;
  __syncthreads();
  if (tid < 16) {
    float mm = part[tid * 8];
#pragma unroll
    for (int i = 1; i < 8; ++i) mm = fmaxf(mm, part[tid * 8 + i]);
    rowmax[tid] = mm;
  }
  __syncthreads();

  const float rm = rowmax[srow];
  float sum = 0.0f;
#pragma unroll 4
  for (int i = 0; i < 32; ++i) {
    const v8h v = *(const v8h*)(Srow + i * 8);
    v8h p;
#pragma unroll
    for (int e = 0; e < 8; ++e) {
      const float pe = __expf((float)v[e] - rm);
      sum += pe;
      p[e] = (_Float16)pe;
    }
    *(v8h*)(Srow + i * 8) = p;          // exp in place (same thread, same slots)
  }
  part[srow * 8 + slane] = sum;
  __syncthreads();
  if (tid < 16) {
    float ss = 0.0f;
#pragma unroll
    for (int i = 0; i < 8; ++i) ss += part[tid * 8 + i];
    rowsum[tid] = ss;
  }
  __syncthreads();

  // ---- Phase 3: O = P @ V ; wave w owns d-tile w ----
  v8f o = {};
  const int dcol = wave * 16 + row16;
  for (int key0 = 0; key0 < T_; key0 += 32) {
    const v16h ap = load16_gap(&S[row16 * SP_ + key0 + (hi ? 8 : 0)]);
    const v16h bv = load16_contig(vb + (size_t)dcol * T_ + key0 + (hi ? 8 : 0));
    o = wmma16(ap, bv, o);
  }

  const int b = bh >> 4, h = bh & 15;
#pragma unroll
  for (int e = 0; e < 8; ++e) {
    const int r = (hi ? 8 : 0) + e;
    const float val = o[e] / rowsum[r];
    const int t = qt * 16 + r;
    attn[((size_t)(b * T_ + t)) * C_ + h * 64 + dcol] = (_Float16)val;
  }
}

// ---------------------------------------------------------------------------
extern "C" void kernel_launch(void* const* d_in, const int* in_sizes, int n_in,
                              void* d_out, int out_size, void* d_ws, size_t ws_size,
                              hipStream_t stream) {
  (void)in_sizes; (void)n_in; (void)out_size; (void)ws_size;
  const float* x    = (const float*)d_in[0];
  const float* Wqkv = (const float*)d_in[1];
  const float* bqkv = (const float*)d_in[2];
  const float* Wout = (const float*)d_in[3];
  const float* bout = (const float*)d_in[4];
  float* out = (float*)d_out;

  char* ws = (char*)d_ws;
  float*    qkv = (float*)ws;                                // 50,331,648 B
  _Float16* qh  = (_Float16*)(ws + (size_t)M_ * 3072 * 4);   //  8,388,608 B
  _Float16* kh  = qh + (size_t)B_ * H_ * T_ * D_;
  _Float16* vt  = kh + (size_t)B_ * H_ * T_ * D_;
  _Float16* att = vt + (size_t)B_ * H_ * T_ * D_;

  // 1) qkv = x @ Wqkv + bqkv
  gemm_wmma_kernel<false><<<dim3(M_ / 128, 3072 / 64), 256, 0, stream>>>(
      x, Wqkv, bqkv, qkv, C_, 3 * C_);

  // 2) RoPE + pack to f16 (q scaled+roped; k roped; v transposed)
  rope_pack_kernel<<<(B_ * H_ * T_ * D_) / 256, 256, 0, stream>>>(qkv, qh, kh, vt);

  // 3) attention
  const int smem_bytes = 16 * SP_ * 2 + (16 * 8 + 32) * 4;   // ~66.4 KB
  hipFuncSetAttribute((const void*)attn_wmma_kernel,
                      hipFuncAttributeMaxDynamicSharedMemorySize, smem_bytes);
  attn_wmma_kernel<<<dim3(B_ * H_ * (T_ / 16)), 128, smem_bytes, stream>>>(
      qh, kh, vt, att);

  // 4) out = attn @ Wout + bout
  gemm_wmma_kernel<true><<<dim3(M_ / 128, C_ / 64), 256, 0, stream>>>(
      att, Wout, bout, out, C_, C_);
}
